// CentroidsTripletLoss_17102559773294
// MI455X (gfx1250) — compile-verified, matrix operand order checked by reading, not run on hardware
//
#include <hip/hip_runtime.h>
#include <hip/hip_bf16.h>
#include <math.h>

typedef __bf16 v16bf __attribute__((ext_vector_type(16)));
typedef __bf16 v8bf  __attribute__((ext_vector_type(8)));
typedef __bf16 v4bf  __attribute__((ext_vector_type(4)));
typedef float  v8f   __attribute__((ext_vector_type(8)));

constexpr int kB = 4096;
constexpr int kD = 1024;
constexpr int kC = 5000;
constexpr int kCP = 5008;                  // padded to 313 full 16-wide tiles
constexpr int kR = 3 * kB;                 // 12288 logit rows
constexpr int kNT = kCP / 16;              // 313 N tiles of 16
constexpr int kNSPLIT = 8;

constexpr size_t kAElems = (size_t)kB * kD;   // 4,194,304
constexpr size_t kEElems = (size_t)kCP * kD;  // 5,128,192
constexpr size_t kFloatBytes = 91136 * sizeof(float);              // base scratch
constexpr size_t kWsNeeded = kFloatBytes + 2 * (kAElems + kEElems) * sizeof(__bf16);

// ---------------------------------------------------------------------------
// Split f32 -> bf16 hi/lo and pack into the 16x16x32 WMMA per-lane layout.
// (fallback path only)
// ---------------------------------------------------------------------------
__device__ __forceinline__ void split_pack(float4 c0, float4 c1, float4 c2, float4 c3,
                                           v16bf& hi, v16bf& lo) {
  float f[16] = {c0.x, c0.y, c0.z, c0.w, c1.x, c1.y, c1.z, c1.w,
                 c2.x, c2.y, c2.z, c2.w, c3.x, c3.y, c3.z, c3.w};
#pragma unroll
  for (int e = 0; e < 16; ++e) {
    __bf16 h = (__bf16)f[e];
    hi[e] = h;
    lo[e] = (__bf16)(f[e] - (float)h);
  }
}

// ---------------------------------------------------------------------------
// Kernel 0: elementwise f32 -> (bf16 hi, bf16 lo) split, with zero padding.
// ---------------------------------------------------------------------------
__global__ void __launch_bounds__(256)
split_cvt_kernel(const float* __restrict__ src, size_t srcElems, size_t totElems,
                 __bf16* __restrict__ hi, __bf16* __restrict__ lo) {
  size_t base = ((size_t)blockIdx.x * 256 + threadIdx.x) * 4;
  if (base >= totElems) return;
  float4 f;
  if (base < srcElems) f = *(const float4*)(src + base);   // both multiples of 4
  else                 f = make_float4(0.f, 0.f, 0.f, 0.f);
  float v[4] = {f.x, f.y, f.z, f.w};
  v4bf h, l;
#pragma unroll
  for (int e = 0; e < 4; ++e) {
    __bf16 hh = (__bf16)v[e];
    h[e] = hh;
    l[e] = (__bf16)(v[e] - (float)hh);
  }
  *(v4bf*)(hi + base) = h;
  *(v4bf*)(lo + base) = l;
}

// ---------------------------------------------------------------------------
// Kernel 1: per anchor row: a2[m] = |a|^2 ; trip[m] = relu(|a-p| - |a-n|)
// ---------------------------------------------------------------------------
__global__ void __launch_bounds__(256)
row_norms_kernel(const float* __restrict__ a, const float* __restrict__ p,
                 const float* __restrict__ ng, float* __restrict__ a2,
                 float* __restrict__ trip) {
  __shared__ float r1[256], r2[256], r3[256];
  const int row = blockIdx.x, tid = threadIdx.x;
  const float* ar = a  + (size_t)row * kD;
  const float* pr = p  + (size_t)row * kD;
  const float* nr = ng + (size_t)row * kD;
  float sap = 0.f, san = 0.f, sa2 = 0.f;
  for (int i = tid; i < kD; i += 256) {
    float av = ar[i];
    float d1 = av - pr[i];
    float d2 = av - nr[i];
    sap += d1 * d1; san += d2 * d2; sa2 += av * av;
  }
  r1[tid] = sap; r2[tid] = san; r3[tid] = sa2;
  __syncthreads();
  for (int s = 128; s > 0; s >>= 1) {
    if (tid < s) { r1[tid] += r1[tid + s]; r2[tid] += r2[tid + s]; r3[tid] += r3[tid + s]; }
    __syncthreads();
  }
  if (tid == 0) {
    trip[row] = fmaxf(sqrtf(r1[0]) - sqrtf(r2[0]), 0.0f);
    a2[row] = r3[0];
  }
}

// ---------------------------------------------------------------------------
// Kernel 2: per exemplar row: e2[c] = |e|^2 ; padded rows get +huge
// ---------------------------------------------------------------------------
__global__ void __launch_bounds__(256)
e2_kernel(const float* __restrict__ ex, float* __restrict__ e2) {
  __shared__ float red[256];
  const int row = blockIdx.x, tid = threadIdx.x;
  if (row >= kC) {
    if (tid == 0) e2[row] = 3.4e38f;
    return;
  }
  const float* er = ex + (size_t)row * kD;
  float s = 0.f;
  for (int i = tid; i < kD; i += 256) { float v = er[i]; s += v * v; }
  red[tid] = s;
  __syncthreads();
  for (int st = 128; st > 0; st >>= 1) {
    if (tid < st) red[tid] += red[tid + st];
    __syncthreads();
  }
  if (tid == 0) e2[row] = red[0];
}

// ---------------------------------------------------------------------------
// Kernel 3: cross-entropy per logit row (values cached in registers)
// ---------------------------------------------------------------------------
__global__ void __launch_bounds__(256)
softmax_kernel(const float* __restrict__ outputs, const int* __restrict__ la,
               const int* __restrict__ ln, float* __restrict__ sm) {
  __shared__ float red[256];
  const int row = blockIdx.x, tid = threadIdx.x;
  const float* x = outputs + (size_t)row * kC;
  float v[20];
  float mx = -3.4e38f;
#pragma unroll
  for (int j = 0; j < 20; ++j) {
    int i = tid + j * 256;
    float f = (i < kC) ? x[i] : -3.4e38f;
    v[j] = f;
    mx = fmaxf(mx, f);
  }
  red[tid] = mx; __syncthreads();
  for (int s = 128; s > 0; s >>= 1) {
    if (tid < s) red[tid] = fmaxf(red[tid], red[tid + s]);
    __syncthreads();
  }
  const float rowmax = red[0];
  __syncthreads();
  float se = 0.f;
#pragma unroll
  for (int j = 0; j < 20; ++j) se += __expf(v[j] - rowmax);   // pad underflows to 0
  red[tid] = se; __syncthreads();
  for (int s = 128; s > 0; s >>= 1) {
    if (tid < s) red[tid] += red[tid + s];
    __syncthreads();
  }
  if (tid == 0) {
    int lab = (row < 2 * kB) ? la[row & (kB - 1)] : ln[row - 2 * kB];
    float xl = x[lab];
    sm[row] = -(xl - rowmax - logf(red[0])) * (1.0f / (float)kR);
  }
}

// ---------------------------------------------------------------------------
// Kernel 4 (fast): fused GEMM on pre-split bf16 hi/lo + distance min / label
// capture. Branch-free inner loop: 8x16B loads + 3 WMMA per K-tile of 32.
// ---------------------------------------------------------------------------
__global__ void __launch_bounds__(256)
center_gemm_bf16_kernel(const __bf16* __restrict__ ahi, const __bf16* __restrict__ alo,
                        const __bf16* __restrict__ ehi, const __bf16* __restrict__ elo,
                        const int* __restrict__ labels, const float* __restrict__ a2,
                        const float* __restrict__ e2,
                        float* __restrict__ dmin_out, float* __restrict__ dref_out) {
  const int lane = threadIdx.x & 31;
  const int wave = threadIdx.x >> 5;
  const int half = lane >> 4;         // K-half selector (A/B layout)
  const int lm   = lane & 15;         // A row / B column within tile
  const int Mbase = blockIdx.x * 128 + wave * 16;
  const int chunk = blockIdx.y;
  const int tiles_per = (kNT + kNSPLIT - 1) / kNSPLIT;
  const int t0 = chunk * tiles_per;
  int t1 = t0 + tiles_per; if (t1 > kNT) t1 = kNT;

  float a2r[8]; int lab[8]; float dmin[8]; float dref[8];
#pragma unroll
  for (int r = 0; r < 8; ++r) {
    int m = Mbase + r + 8 * half;     // C/D layout: lane,VGPR r -> row M
    a2r[r] = a2[m];
    lab[r] = labels[m];
    dmin[r] = 3.4e38f;
    dref[r] = 0.0f;
  }

  const __bf16* Ahi = ahi + (size_t)(Mbase + lm) * kD;
  const __bf16* Alo = alo + (size_t)(Mbase + lm) * kD;
  const int koff = 8 * half;

  for (int t = t0; t < t1; ++t) {
    const int n = t * 16 + lm;        // exemplar column (always < kCP, padded)
    const __bf16* Bhi = ehi + (size_t)n * kD;
    const __bf16* Blo = elo + (size_t)n * kD;
    v8f acc = {};
#pragma unroll 2
    for (int k0 = 0; k0 < kD; k0 += 32) {
      const int o1 = k0 + koff;       // K = 8*half + e      (e in [0,8))
      const int o2 = k0 + 16 + koff;  // K = 16 + 8*half + e (e in [8,16))
      v8bf ah0 = *(const v8bf*)(Ahi + o1);
      v8bf ah1 = *(const v8bf*)(Ahi + o2);
      v8bf al0 = *(const v8bf*)(Alo + o1);
      v8bf al1 = *(const v8bf*)(Alo + o2);
      v8bf bh0 = *(const v8bf*)(Bhi + o1);
      v8bf bh1 = *(const v8bf*)(Bhi + o2);
      v8bf bl0 = *(const v8bf*)(Blo + o1);
      v8bf bl1 = *(const v8bf*)(Blo + o2);
      v16bf Ah = __builtin_shufflevector(ah0, ah1, 0,1,2,3,4,5,6,7,8,9,10,11,12,13,14,15);
      v16bf Al = __builtin_shufflevector(al0, al1, 0,1,2,3,4,5,6,7,8,9,10,11,12,13,14,15);
      v16bf Bh = __builtin_shufflevector(bh0, bh1, 0,1,2,3,4,5,6,7,8,9,10,11,12,13,14,15);
      v16bf Bl = __builtin_shufflevector(bl0, bl1, 0,1,2,3,4,5,6,7,8,9,10,11,12,13,14,15);
      // bf16x3: hi*hi + hi*lo + lo*hi ~ f32 dot product
      acc = __builtin_amdgcn_wmma_f32_16x16x32_bf16(false, Ah, false, Bh, (short)0, acc, false, false);
      acc = __builtin_amdgcn_wmma_f32_16x16x32_bf16(false, Ah, false, Bl, (short)0, acc, false, false);
      acc = __builtin_amdgcn_wmma_f32_16x16x32_bf16(false, Al, false, Bh, (short)0, acc, false, false);
    }
    const float e2n = e2[n];          // padded entries hold 3.4e38
#pragma unroll
    for (int r = 0; r < 8; ++r) {
      float sq = a2r[r] + e2n - 2.0f * acc[r];
      float d = sqrtf(fmaxf(sq, 0.0f));
      dmin[r] = fminf(dmin[r], d);
      dref[r] += (n == lab[r]) ? d : 0.0f;
    }
  }

#pragma unroll
  for (int r = 0; r < 8; ++r) {
    float dm = dmin[r], dr = dref[r];
#pragma unroll
    for (int msk = 8; msk >= 1; msk >>= 1) {
      dm = fminf(dm, __shfl_xor(dm, msk, 32));
      dr += __shfl_xor(dr, msk, 32);
    }
    dmin[r] = dm; dref[r] = dr;
  }
  if (lm == 0) {
#pragma unroll
    for (int r = 0; r < 8; ++r) {
      int m = Mbase + r + 8 * half;
      dmin_out[(size_t)chunk * kB + m] = dmin[r];
      dref_out[(size_t)chunk * kB + m] = dref[r];
    }
  }
}

// ---------------------------------------------------------------------------
// Kernel 4 (fallback, small ws): fused conversion variant (round-1 kernel)
// ---------------------------------------------------------------------------
__global__ void __launch_bounds__(256)
center_gemm_kernel(const float* __restrict__ anchor, const float* __restrict__ exem,
                   const int* __restrict__ labels, const float* __restrict__ a2,
                   const float* __restrict__ e2,
                   float* __restrict__ dmin_out, float* __restrict__ dref_out) {
  const int lane = threadIdx.x & 31;
  const int wave = threadIdx.x >> 5;
  const int half = lane >> 4;
  const int lm   = lane & 15;
  const int Mbase = blockIdx.x * 128 + wave * 16;
  const int chunk = blockIdx.y;
  const int tiles_per = (kNT + kNSPLIT - 1) / kNSPLIT;
  const int t0 = chunk * tiles_per;
  int t1 = t0 + tiles_per; if (t1 > kNT) t1 = kNT;

  float a2r[8]; int lab[8]; float dmin[8]; float dref[8];
#pragma unroll
  for (int r = 0; r < 8; ++r) {
    int m = Mbase + r + 8 * half;
    a2r[r] = a2[m];
    lab[r] = labels[m];
    dmin[r] = 3.4e38f;
    dref[r] = 0.0f;
  }

  const float* Arow = anchor + (size_t)(Mbase + lm) * kD;

  for (int t = t0; t < t1; ++t) {
    const int n = t * 16 + lm;
    const bool nv = (n < kC);
    const float* Brow = exem + (size_t)(nv ? n : 0) * kD;
    v8f acc = {};
    for (int k0 = 0; k0 < kD; k0 += 32) {
      const float* ap = Arow + k0 + 8 * half;
      float4 ax0 = *(const float4*)(ap);
      float4 ax1 = *(const float4*)(ap + 4);
      float4 ax2 = *(const float4*)(ap + 16);
      float4 ax3 = *(const float4*)(ap + 20);
      float4 bx0, bx1, bx2, bx3;
      const float* bp = Brow + k0 + 8 * half;
      if (nv) {
        bx0 = *(const float4*)(bp);
        bx1 = *(const float4*)(bp + 4);
        bx2 = *(const float4*)(bp + 16);
        bx3 = *(const float4*)(bp + 20);
      } else {
        bx0 = bx1 = bx2 = bx3 = make_float4(0.f, 0.f, 0.f, 0.f);
      }
      v16bf ahiV, aloV, bhiV, bloV;
      split_pack(ax0, ax1, ax2, ax3, ahiV, aloV);
      split_pack(bx0, bx1, bx2, bx3, bhiV, bloV);
      acc = __builtin_amdgcn_wmma_f32_16x16x32_bf16(false, ahiV, false, bhiV, (short)0, acc, false, false);
      acc = __builtin_amdgcn_wmma_f32_16x16x32_bf16(false, ahiV, false, bloV, (short)0, acc, false, false);
      acc = __builtin_amdgcn_wmma_f32_16x16x32_bf16(false, aloV, false, bhiV, (short)0, acc, false, false);
    }
    const float e2n = nv ? e2[n] : 3.4e38f;
#pragma unroll
    for (int r = 0; r < 8; ++r) {
      float sq = a2r[r] + e2n - 2.0f * acc[r];
      float d = sqrtf(fmaxf(sq, 0.0f));
      dmin[r] = fminf(dmin[r], d);
      dref[r] += (n == lab[r]) ? d : 0.0f;
    }
  }
#pragma unroll
  for (int r = 0; r < 8; ++r) {
    float dm = dmin[r], dr = dref[r];
#pragma unroll
    for (int msk = 8; msk >= 1; msk >>= 1) {
      dm = fminf(dm, __shfl_xor(dm, msk, 32));
      dr += __shfl_xor(dr, msk, 32);
    }
    dmin[r] = dm; dref[r] = dr;
  }
  if (lm == 0) {
#pragma unroll
    for (int r = 0; r < 8; ++r) {
      int m = Mbase + r + 8 * half;
      dmin_out[(size_t)chunk * kB + m] = dmin[r];
      dref_out[(size_t)chunk * kB + m] = dref[r];
    }
  }
}

// ---------------------------------------------------------------------------
// Kernel 5: deterministic final reduction + combine
// ---------------------------------------------------------------------------
__global__ void __launch_bounds__(1024)
final_kernel(const float* __restrict__ sm, const float* __restrict__ trip,
             const float* __restrict__ dminp, const float* __restrict__ drefp,
             float* __restrict__ out) {
  __shared__ float red[1024];
  const int tid = threadIdx.x;
  float ssm = 0.f, str = 0.f, sce = 0.f;
  for (int i = tid; i < kR; i += 1024) ssm += sm[i];
  for (int i = tid; i < kB; i += 1024) str += trip[i];
  for (int m = tid; m < kB; m += 1024) {
    float dm = 3.4e38f, dr = 0.f;
#pragma unroll
    for (int c = 0; c < kNSPLIT; ++c) {
      dm = fminf(dm, dminp[(size_t)c * kB + m]);
      dr += drefp[(size_t)c * kB + m];
    }
    sce += fmaxf(dr - dm, 0.0f);
  }
  float sums[3] = {ssm, str, sce};
  float res[3];
#pragma unroll
  for (int k = 0; k < 3; ++k) {
    red[tid] = sums[k];
    __syncthreads();
    for (int s = 512; s > 0; s >>= 1) {
      if (tid < s) red[tid] += red[tid + s];
      __syncthreads();
    }
    res[k] = red[0];
    __syncthreads();
  }
  if (tid == 0) {
    const float softmax = res[0], triplet = res[1], center = res[2];
    out[0] = softmax + 1.0f * triplet + 0.5f * center;  // ALPHA=1, BETA=0.5
    out[1] = triplet;
    out[2] = softmax;
    out[3] = center;
  }
}

// ---------------------------------------------------------------------------
extern "C" void kernel_launch(void* const* d_in, const int* in_sizes, int n_in,
                              void* d_out, int out_size, void* d_ws, size_t ws_size,
                              hipStream_t stream) {
  const float* anchor   = (const float*)d_in[0];
  const float* positive = (const float*)d_in[1];
  const float* negative = (const float*)d_in[2];
  const float* outputs  = (const float*)d_in[3];
  const int*   la       = (const int*)d_in[4];
  const int*   ln       = (const int*)d_in[5];
  const float* exem     = (const float*)d_in[6];
  float* out = (float*)d_out;

  float* ws    = (float*)d_ws;
  float* a2    = ws;                 // [4096]
  float* e2    = ws + 4096;          // [5008] (padded)
  float* sm    = ws + 9216;          // [12288]
  float* trip  = ws + 21504;         // [4096]
  float* dminp = ws + 25600;         // [8][4096]
  float* drefp = ws + 58368;         // [8][4096]  (floats end at 91136)

  row_norms_kernel<<<kB, 256, 0, stream>>>(anchor, positive, negative, a2, trip);
  e2_kernel<<<kCP, 256, 0, stream>>>(exem, e2);
  softmax_kernel<<<kR, 256, 0, stream>>>(outputs, la, ln, sm);

  dim3 gg(kB / 128, kNSPLIT, 1);
  if (ws_size >= kWsNeeded) {
    __bf16* ahi = (__bf16*)((char*)d_ws + kFloatBytes);
    __bf16* alo = ahi + kAElems;
    __bf16* ehi = alo + kAElems;
    __bf16* elo = ehi + kEElems;
    {
      size_t g = (kAElems / 4 + 255) / 256;
      split_cvt_kernel<<<(unsigned)g, 256, 0, stream>>>(anchor, kAElems, kAElems, ahi, alo);
    }
    {
      size_t g = (kEElems / 4 + 255) / 256;
      split_cvt_kernel<<<(unsigned)g, 256, 0, stream>>>(exem, (size_t)kC * kD, kEElems, ehi, elo);
    }
    center_gemm_bf16_kernel<<<gg, 256, 0, stream>>>(ahi, alo, ehi, elo, la, a2, e2,
                                                    dminp, drefp);
  } else {
    center_gemm_kernel<<<gg, 256, 0, stream>>>(anchor, exem, la, a2, e2, dminp, drefp);
  }
  final_kernel<<<1, 1024, 0, stream>>>(sm, trip, dminp, drefp, out);
}
